// MultiHeadAttention_20169166422807
// MI455X (gfx1250) — matrix operand order repaired and checked by reading, not compiled, not run on hardware
//
#include <hip/hip_runtime.h>
#include <hip/hip_bf16.h>

// ---------------------------------------------------------------------------
// MultiHeadAttention forward for MI455X (gfx1250), wave32 + WMMA bf16 + TDM.
//   x:(4,2048,1024) f32, mask:(4,2048) bool, Wqkv:(3072,1024) f32,
//   Wout:(1024,1024) f32, bout:(1024) f32  ->  out:(4,2048,1024) f32
// Pipeline: f32->bf16 convert, qkv GEMM (WMMA, TDM double-buffered),
// flash attention (WMMA, TDM K staging), out projection GEMM (WMMA, +bias).
// ---------------------------------------------------------------------------

#define DIMV     1024
#define HEADSV   16
#define HDIM     64
#define BATCHV   4
#define SEQV     2048
#define BSV      (BATCHV * SEQV)   // 8192
#define NQKV     (3 * DIMV)        // 3072

typedef __attribute__((ext_vector_type(16))) __bf16 v16bf;
typedef __attribute__((ext_vector_type(8)))  float  v8f;
typedef __attribute__((ext_vector_type(4)))  unsigned int v4u;

union FragU { v16bf v; v4u u[2]; };

// ---- Tensor Data Mover availability (device pass only, probe-guarded) -----
#if defined(__HIP_DEVICE_COMPILE__) && defined(__has_builtin)
#if __has_builtin(__builtin_amdgcn_tensor_load_to_lds) && \
    __has_builtin(__builtin_amdgcn_s_wait_tensorcnt)
#define USE_TDM 1
#endif
#endif
#ifndef USE_TDM
#define USE_TDM 0
#endif

#if USE_TDM
typedef unsigned int u32x4 __attribute__((ext_vector_type(4)));
typedef int          i32x4 __attribute__((ext_vector_type(4)));
typedef int          i32x8 __attribute__((ext_vector_type(8)));

// DMA a (rows x 64) bf16 tile, row stride `stride_elems`, into LDS with rows
// padded to 72 elements (pad_interval = 32 DWORDs, pad_amount = 4 DWORDs).
// D# built per CDNA5 ISA ch.8 (group0: ctrl/lds/global/type; group1: dims).
__device__ __forceinline__ void tdm_load_2d(const unsigned short* gsrc,
                                            unsigned short* ldst,
                                            int rows, int stride_elems) {
  const unsigned long long ga = (unsigned long long)(const void*)gsrc;
  const unsigned int lds = (unsigned int)(unsigned long long)(void*)ldst;
  u32x4 g0;
  g0[0] = 1u;                                               // count=1 (valid)
  g0[1] = lds;                                              // lds_addr
  g0[2] = (unsigned int)ga;                                 // global_addr lo
  g0[3] = (unsigned int)((ga >> 32) & 0x1ffffffull) | (2u << 30);  // hi|type=2
  i32x8 g1;
  g1[0] = (1 << 16) | (1 << 20) | (4 << 22) | (3 << 25);    // 2B, pad 4dw/32dw
  g1[1] = (64 << 16);                                       // tensor_dim0=64
  g1[2] = ((rows & 0xffff) << 16);                          // tensor_dim1
  g1[3] = (64 << 16);                                       // tile_dim0=64
  g1[4] = (rows & 0xffff);                                  // tile_dim1
  g1[5] = stride_elems;                                     // dim0 stride lo
  g1[6] = 0;
  g1[7] = 0;
  const i32x4 gz = {0, 0, 0, 0};
#if __clang_major__ >= 23
  const i32x8 gz8 = {0, 0, 0, 0, 0, 0, 0, 0};
  __builtin_amdgcn_tensor_load_to_lds(g0, g1, gz, gz, gz8, 0);
#else
  __builtin_amdgcn_tensor_load_to_lds(g0, g1, gz, gz, 0);
#endif
}
#endif  // USE_TDM

__device__ __forceinline__ unsigned short f2bf(float f) {
  unsigned int u = __float_as_uint(f);
  u += 0x7fffu + ((u >> 16) & 1u);          // round-to-nearest-even
  return (unsigned short)(u >> 16);
}

__device__ __forceinline__ v8f wmma_bf16(v16bf a, v16bf b, v8f c) {
  return __builtin_amdgcn_wmma_f32_16x16x32_bf16(
      false, a, false, b, (short)0, c, false, false);
}

// A-operand 16x32 bf16 frag from row-major [row][k] storage (global or LDS).
__device__ __forceinline__ v16bf load_a_frag(const unsigned short* base,
                                             int ld, int lane) {
  const int r = lane & 15;
  const int h = (lane >> 4) & 1;
  const unsigned short* p = base + (size_t)r * ld + 8 * h;
  FragU f;
  f.u[0] = *(const v4u*)(p);
  f.u[1] = *(const v4u*)(p + 16);
  return f.v;
}

// B-operand 32x16 bf16 frag from [n][k] storage (column n contiguous in k).
__device__ __forceinline__ v16bf load_b_frag(const unsigned short* base,
                                             int ld, int lane) {
  const int n = lane & 15;
  const int h = (lane >> 4) & 1;
  const unsigned short* p = base + (size_t)n * ld + 8 * h;
  FragU f;
  f.u[0] = *(const v4u*)(p);
  f.u[1] = *(const v4u*)(p + 16);
  return f.v;
}

// Manual fallback: stage a (128 x 64) bf16 tile into LDS (rows padded to 72).
__device__ __forceinline__ void stage_tile_manual(
    const unsigned short* __restrict__ G, int ld, int tid,
    unsigned short (*dst)[72]) {
#pragma unroll
  for (int cc = 0; cc < 2; ++cc) {
    const int c = tid * 2 + cc;          // 0..511
    const int r = c >> 2, off = (c & 3) * 16;
    const unsigned short* g = G + (size_t)r * ld + off;
    *(v4u*)&dst[r][off]     = *(const v4u*)(g);
    *(v4u*)&dst[r][off + 8] = *(const v4u*)(g + 8);
  }
}

// ---------------------------------------------------------------------------
__global__ __launch_bounds__(256)
void f32_to_bf16_kernel(const float* __restrict__ in,
                        unsigned short* __restrict__ out, int n) {
  int i = blockIdx.x * blockDim.x + threadIdx.x;
  int stride = gridDim.x * blockDim.x;
  for (; i < n; i += stride) out[i] = f2bf(in[i]);
}

// ---------------------------------------------------------------------------
// C[M,N] = A[M,K] * W[N,K]^T   (bf16 operands, f32 accumulate)
// EPI==0: bf16 out, scale cols < DIMV by 0.125 (Q pre-scale for attention)
// EPI==1: f32 out, +bias
// Block tile 128x128, 256 threads = 8 waves (4 M x 2 N), wave tile 32x64,
// K staged 64-wide, double-buffered via TDM (wave 0 issues DMA for tile t+1
// while all waves compute tile t; TENSORcnt drained before the barrier).
// ---------------------------------------------------------------------------
template <int EPI>
__global__ __launch_bounds__(256)
void gemm_bf16_kernel(const unsigned short* __restrict__ A,
                      const unsigned short* __restrict__ W,
                      unsigned short* __restrict__ Cb,
                      float* __restrict__ Cf,
                      const float* __restrict__ bias,
                      int M, int N, int K) {
  __shared__ unsigned short Asl[2][128][72];
  __shared__ unsigned short Bsl[2][128][72];

  const int tid  = threadIdx.x;
  const int lane = tid & 31;
  const int wave = tid >> 5;      // 0..7
  const int wm   = wave >> 1;     // 0..3  (M)
  const int wn   = wave & 1;      // 0..1  (N)
  const int m_base = blockIdx.y * 128;
  const int n_base = blockIdx.x * 128;

  v8f acc[2][4];
  const v8f vz = {0.f, 0.f, 0.f, 0.f, 0.f, 0.f, 0.f, 0.f};
#pragma unroll
  for (int i = 0; i < 2; ++i)
#pragma unroll
    for (int j = 0; j < 4; ++j) acc[i][j] = vz;

  auto stage = [&](int buf, int k0) {
#if USE_TDM
    if (wave == 0) {
      tdm_load_2d(A + (size_t)m_base * K + k0, &Asl[buf][0][0], 128, K);
      tdm_load_2d(W + (size_t)n_base * K + k0, &Bsl[buf][0][0], 128, K);
    }
#else
    stage_tile_manual(A + (size_t)m_base * K + k0, K, tid, Asl[buf]);
    stage_tile_manual(W + (size_t)n_base * K + k0, K, tid, Bsl[buf]);
#endif
  };

  stage(0, 0);
#if USE_TDM
  if (wave == 0) __builtin_amdgcn_s_wait_tensorcnt(0);
#endif
  __syncthreads();

  const int nk = K / 64;
  for (int kt = 0; kt < nk; ++kt) {
    const int cur = kt & 1;
    if (kt + 1 < nk) stage(cur ^ 1, (kt + 1) * 64);   // prefetch next tile
#pragma unroll
    for (int ks = 0; ks < 2; ++ks) {
      const v16bf a0 = load_a_frag(&Asl[cur][wm * 32][ks * 32],      72, lane);
      const v16bf a1 = load_a_frag(&Asl[cur][wm * 32 + 16][ks * 32], 72, lane);
#pragma unroll
      for (int jn = 0; jn < 4; ++jn) {
        const v16bf bb =
            load_b_frag(&Bsl[cur][wn * 64 + jn * 16][ks * 32], 72, lane);
        acc[0][jn] = wmma_bf16(a0, bb, acc[0][jn]);
        acc[1][jn] = wmma_bf16(a1, bb, acc[1][jn]);
      }
    }
#if USE_TDM
    if (wave == 0) __builtin_amdgcn_s_wait_tensorcnt(0);
#endif
    __syncthreads();
  }

  // C/D layout: lane holds col n=lane&15, rows (lane>>4)*8 + i
  const int cn = lane & 15;
  const int mo = (lane >> 4) * 8;
#pragma unroll
  for (int im = 0; im < 2; ++im) {
#pragma unroll
    for (int jn = 0; jn < 4; ++jn) {
      const int row0 = m_base + wm * 32 + im * 16 + mo;
      const int col  = n_base + wn * 64 + jn * 16 + cn;
      if (EPI == 0) {
        const float s = (col < DIMV) ? 0.125f : 1.0f;   // fold 1/sqrt(64)
#pragma unroll
        for (int i = 0; i < 8; ++i)
          Cb[(size_t)(row0 + i) * N + col] = f2bf(acc[im][jn][i] * s);
      } else {
        const float bv = bias[col];
#pragma unroll
        for (int i = 0; i < 8; ++i)
          Cf[(size_t)(row0 + i) * N + col] = acc[im][jn][i] + bv;
      }
    }
  }
}

// ---------------------------------------------------------------------------
// Flash attention. Grid (S/64, HEADS, BATCH), 128 threads = 4 waves.
// Wave w owns 16 query rows; streams 64-key blocks. K tile DMA'd by the TDM
// while V is manually transposed into LDS; Q pre-scaled by 0.125.
// ---------------------------------------------------------------------------
__global__ __launch_bounds__(128)
void flash_attn_kernel(const unsigned short* __restrict__ qkv,
                       const unsigned char* __restrict__ mask,
                       unsigned short* __restrict__ obuf) {
  __shared__ unsigned short Kl[64][72];       // [key][d]
  __shared__ unsigned short Vt[64][72];       // [d][key]
  __shared__ unsigned short Pl[4][16][72];    // per-wave P (C->A relayout)

  const int tid  = threadIdx.x;
  const int lane = tid & 31;
  const int wave = tid >> 5;                  // 0..3
  const int qt = blockIdx.x, h = blockIdx.y, b = blockIdx.z;

  const int cn = lane & 15;
  const int mo = ((lane >> 4) & 1) * 8;

  const size_t qrow0 = (size_t)(b * SEQV + qt * 64 + wave * 16);

  v16bf qf[2];
#pragma unroll
  for (int ks = 0; ks < 2; ++ks)
    qf[ks] = load_a_frag(qkv + qrow0 * NQKV + h * HDIM + ks * 32, NQKV, lane);

  const v8f vz = {0.f, 0.f, 0.f, 0.f, 0.f, 0.f, 0.f, 0.f};
  v8f o[4];
#pragma unroll
  for (int dt = 0; dt < 4; ++dt) o[dt] = vz;
  float mrow[8], lrow[8];
#pragma unroll
  for (int i = 0; i < 8; ++i) { mrow[i] = -1e30f; lrow[i] = 0.f; }

  for (int kb = 0; kb < SEQV / 64; ++kb) {
    const int kk0 = kb * 64;
    // ---- stage K block (64 x 64): TDM if available, else manual
#if USE_TDM
    if (wave == 0)
      tdm_load_2d(qkv + (size_t)(b * SEQV + kk0) * NQKV + DIMV + h * HDIM,
                  &Kl[0][0], 64, NQKV);
#else
#pragma unroll
    for (int cc = 0; cc < 2; ++cc) {
      const int c = tid * 2 + cc;
      const int kk = c >> 2, off = (c & 3) * 16;
      const unsigned short* g =
          qkv + (size_t)(b * SEQV + kk0 + kk) * NQKV + DIMV + h * HDIM + off;
      *(v4u*)&Kl[kk][off]     = *(const v4u*)(g);
      *(v4u*)&Kl[kk][off + 8] = *(const v4u*)(g + 8);
    }
#endif
    // ---- stage V transposed (overlaps the K DMA)
    {
      const int kk = tid >> 1, d0 = (tid & 1) * 32;
      const unsigned short* g =
          qkv + (size_t)(b * SEQV + kk0 + kk) * NQKV + 2 * DIMV + h * HDIM + d0;
      unsigned short buf[32];
      *(v4u*)&buf[0]  = *(const v4u*)(g);
      *(v4u*)&buf[8]  = *(const v4u*)(g + 8);
      *(v4u*)&buf[16] = *(const v4u*)(g + 16);
      *(v4u*)&buf[24] = *(const v4u*)(g + 24);
#pragma unroll
      for (int j = 0; j < 32; ++j) Vt[d0 + j][kk] = buf[j];
    }
#if USE_TDM
    if (wave == 0) __builtin_amdgcn_s_wait_tensorcnt(0);
#endif
    __syncthreads();

    // ---- scores S = Q K^T, 4 tiles of 16 keys
    v8f s[4];
#pragma unroll
    for (int nt = 0; nt < 4; ++nt) {
      s[nt] = vz;
#pragma unroll
      for (int ks = 0; ks < 2; ++ks)
        s[nt] = wmma_bf16(qf[ks], load_b_frag(&Kl[nt * 16][ks * 32], 72, lane),
                          s[nt]);
    }

    // ---- key mask
#pragma unroll
    for (int nt = 0; nt < 4; ++nt) {
      const unsigned int mk = mask[(size_t)b * SEQV + kk0 + nt * 16 + cn];
      if (!mk) {
#pragma unroll
        for (int i = 0; i < 8; ++i) s[nt][i] = -1e30f;
      }
    }

    // ---- online softmax (row reductions across 16 lanes, wave32)
    float mloc[8];
#pragma unroll
    for (int i = 0; i < 8; ++i)
      mloc[i] = fmaxf(fmaxf(s[0][i], s[1][i]), fmaxf(s[2][i], s[3][i]));
#pragma unroll
    for (int off = 1; off < 16; off <<= 1)
#pragma unroll
      for (int i = 0; i < 8; ++i)
        mloc[i] = fmaxf(mloc[i], __shfl_xor(mloc[i], off, 16));

    float p[4][8];
#pragma unroll
    for (int i = 0; i < 8; ++i) {
      const float mn = fmaxf(mrow[i], mloc[i]);
      const float sc = __expf(mrow[i] - mn);
      mrow[i] = mn;
      float rs = 0.f;
#pragma unroll
      for (int nt = 0; nt < 4; ++nt) {
        p[nt][i] = __expf(s[nt][i] - mn);
        rs += p[nt][i];
      }
#pragma unroll
      for (int off = 1; off < 16; off <<= 1) rs += __shfl_xor(rs, off, 16);
      lrow[i] = lrow[i] * sc + rs;
#pragma unroll
      for (int dt = 0; dt < 4; ++dt) o[dt][i] *= sc;
    }

    // ---- C-layout -> A-layout via per-wave LDS (LDS in-order per wave)
#pragma unroll
    for (int nt = 0; nt < 4; ++nt)
#pragma unroll
      for (int i = 0; i < 8; ++i)
        Pl[wave][mo + i][nt * 16 + cn] = f2bf(p[nt][i]);

    v16bf pa[2];
#pragma unroll
    for (int ks = 0; ks < 2; ++ks)
      pa[ks] = load_a_frag(&Pl[wave][0][ks * 32], 72, lane);

    // ---- O += P V  (B operand from transposed V)
#pragma unroll
    for (int dt = 0; dt < 4; ++dt)
#pragma unroll
      for (int ks = 0; ks < 2; ++ks)
        o[dt] = wmma_bf16(pa[ks], load_b_frag(&Vt[dt * 16][ks * 32], 72, lane),
                          o[dt]);

    __syncthreads();   // protect Kl/Vt before next block's staging
  }

  float inv[8];
#pragma unroll
  for (int i = 0; i < 8; ++i) inv[i] = 1.0f / fmaxf(lrow[i], 1e-30f);
#pragma unroll
  for (int dt = 0; dt < 4; ++dt)
#pragma unroll
    for (int i = 0; i < 8; ++i)
      obuf[(qrow0 + mo + i) * DIMV + h * HDIM + dt * 16 + cn] =
          f2bf(o[dt][i] * inv[i]);
}

// ---------------------------------------------------------------------------
extern "C" void kernel_launch(void* const* d_in, const int* in_sizes, int n_in,
                              void* d_out, int out_size, void* d_ws,
                              size_t ws_size, hipStream_t stream) {
  (void)in_sizes; (void)n_in; (void)out_size; (void)ws_size;
  const float*         x    = (const float*)d_in[0];
  const unsigned char* mask = (const unsigned char*)d_in[1];
  const float*         Wqkv = (const float*)d_in[2];
  const float*         Wout = (const float*)d_in[3];
  const float*         bout = (const float*)d_in[4];
  float*               out  = (float*)d_out;

  char* ws = (char*)d_ws;
  // workspace: [0,16M) xb (reused as obuf) | [16M,22M) Wqkv bf16 |
  //            [22M,24M) Wout bf16 | [24M,72M) qkv bf16 [8192][3072]
  unsigned short* xb    = (unsigned short*)(ws);
  unsigned short* wqkvb = (unsigned short*)(ws + 16777216);
  unsigned short* woutb = (unsigned short*)(ws + 16777216 + 6291456);
  unsigned short* qkvb  = (unsigned short*)(ws + 16777216 + 6291456 + 2097152);
  unsigned short* obuf  = xb;   // xb only consumed by the earlier qkv GEMM

  f32_to_bf16_kernel<<<2048, 256, 0, stream>>>(x, xb, BSV * DIMV);
  f32_to_bf16_kernel<<<1024, 256, 0, stream>>>(Wqkv, wqkvb, NQKV * DIMV);
  f32_to_bf16_kernel<<<512, 256, 0, stream>>>(Wout, woutb, DIMV * DIMV);

  gemm_bf16_kernel<0><<<dim3(NQKV / 128, BSV / 128), 256, 0, stream>>>(
      xb, wqkvb, qkvb, nullptr, nullptr, BSV, NQKV, DIMV);

  flash_attn_kernel<<<dim3(SEQV / 64, HEADSV, BATCHV), 128, 0, stream>>>(
      qkvb, mask, obuf);

  gemm_bf16_kernel<1><<<dim3(DIMV / 128, BSV / 128), 256, 0, stream>>>(
      obuf, woutb, nullptr, out, bout, BSV, DIMV, DIMV);
}
